// MultiheadSparseAttn_18476949307510
// MI455X (gfx1250) — matrix operand order repaired and checked by reading, not compiled, not run on hardware
//
#include <hip/hip_runtime.h>
#include <hip/hip_bf16.h>

typedef __attribute__((ext_vector_type(16))) _Float16 v16h;
typedef __attribute__((ext_vector_type(8)))  float    v8f;

#define LDSTR 72   // halves per LDS tile row: 64 data + 8 pad; 144B rows (16B aligned)
#define TILEH (64 * LDSTR)

union FragU  { v16h v; uint4 q[2]; };
union Pack4  { _Float16 h[4]; uint2 u; };
union Pack8  { _Float16 h[8]; uint4 u; };

// A-matrix 16x32 f16 fragment (ISA 7.12.2):
//  lane m = lane&15 (row), hi = lane>>4
//  halves 0..7  -> K = kBase + hi*8 + (0..7)
//  halves 8..15 -> K = kBase + 16 + hi*8 + (0..7)
__device__ __forceinline__ v16h ld_fragA(const _Float16* tile, int rowBase, int kBase, int lane) {
  const int m  = lane & 15;
  const int hi = lane >> 4;
  const _Float16* p = tile + (rowBase + m) * LDSTR + kBase + hi * 8;
  FragU f;
  f.q[0] = *reinterpret_cast<const uint4*>(p);
  f.q[1] = *reinterpret_cast<const uint4*>(p + 16);
  return f.v;
}

// B-matrix 32x16 f16 fragment:
//  lane n = lane&15 (column), hi = lane>>4
//  halves 0..7 -> K = kBase + hi*8 + (0..7); halves 8..15 -> K = kBase + 16 + hi*8 + (0..7)
// Tile stored row-major with row = B-column index, col = K index.
__device__ __forceinline__ v16h ld_fragB(const _Float16* tile, int rowBase, int kBase, int lane) {
  const int n  = lane & 15;
  const int hi = lane >> 4;
  const _Float16* p = tile + (rowBase + n) * LDSTR + kBase + hi * 8;
  FragU f;
  f.q[0] = *reinterpret_cast<const uint4*>(p);
  f.q[1] = *reinterpret_cast<const uint4*>(p + 16);
  return f.v;
}

// ---- pipelined tile movement: global -> regs, regs -> LDS f16 ----

// K (row-major): thread's share is 8 linear float4 units (idx = row*16 + c4).
__device__ __forceinline__ void fetch_tileK(float4* r, const float* src, int tid) {
  const float4* s = reinterpret_cast<const float4*>(src);
#pragma unroll
  for (int i = 0; i < 8; ++i) r[i] = s[tid + i * 128];
}

__device__ __forceinline__ void store_tileK(_Float16* dst, const float4* r, int tid, float scl) {
#pragma unroll
  for (int i = 0; i < 8; ++i) {
    int idx = tid + i * 128;
    int row = idx >> 4;
    int c4  = idx & 15;
    Pack4 pk;
    pk.h[0] = (_Float16)(r[i].x * scl);
    pk.h[1] = (_Float16)(r[i].y * scl);
    pk.h[2] = (_Float16)(r[i].z * scl);
    pk.h[3] = (_Float16)(r[i].w * scl);
    *reinterpret_cast<uint2*>(dst + row * LDSTR + c4 * 4) = pk.u;
  }
}

// V (to be transposed): thread's share is 2 units of (key-quad, d-quad) = 8 float4.
__device__ __forceinline__ void fetch_tileV(float4* r, const float* src, int tid) {
  const float4* s = reinterpret_cast<const float4*>(src);
#pragma unroll
  for (int i = 0; i < 2; ++i) {
    int u  = tid + i * 128;
    int kq = u >> 4;
    int c4 = u & 15;
#pragma unroll
    for (int j = 0; j < 4; ++j) r[i * 4 + j] = s[(kq * 4 + j) * 16 + c4];
  }
}

__device__ __forceinline__ void store_tileV(_Float16* dst, const float4* r, int tid) {
#pragma unroll
  for (int i = 0; i < 2; ++i) {
    int u  = tid + i * 128;
    int kq = u >> 4;
    int c4 = u & 15;
    const float4 w0 = r[i * 4 + 0], w1 = r[i * 4 + 1], w2 = r[i * 4 + 2], w3 = r[i * 4 + 3];
    Pack4 p0, p1, p2, p3;
    p0.h[0]=(_Float16)w0.x; p0.h[1]=(_Float16)w1.x; p0.h[2]=(_Float16)w2.x; p0.h[3]=(_Float16)w3.x;
    p1.h[0]=(_Float16)w0.y; p1.h[1]=(_Float16)w1.y; p1.h[2]=(_Float16)w2.y; p1.h[3]=(_Float16)w3.y;
    p2.h[0]=(_Float16)w0.z; p2.h[1]=(_Float16)w1.z; p2.h[2]=(_Float16)w2.z; p2.h[3]=(_Float16)w3.z;
    p3.h[0]=(_Float16)w0.w; p3.h[1]=(_Float16)w1.w; p3.h[2]=(_Float16)w2.w; p3.h[3]=(_Float16)w3.w;
    *reinterpret_cast<uint2*>(dst + (c4 * 4 + 0) * LDSTR + kq * 4) = p0.u;
    *reinterpret_cast<uint2*>(dst + (c4 * 4 + 1) * LDSTR + kq * 4) = p1.u;
    *reinterpret_cast<uint2*>(dst + (c4 * 4 + 2) * LDSTR + kq * 4) = p2.u;
    *reinterpret_cast<uint2*>(dst + (c4 * 4 + 3) * LDSTR + kq * 4) = p3.u;
  }
}

// Grid: x = query block (0..63), y = b*H+h. 128 threads = 4 waves, wave owns 16 queries.
// Computes S^T = K Q^T (lane owns one query column) with double-buffered K/V tiles.
// LDS layout (tiles of 64*LDSTR halves):
//   [0] Qs  [1] Ks buf0  [2] Ks buf1  [3] Vt buf0  [4] Vt buf1  [5] Ps (4 x 16 x LDSTR)
__global__ __launch_bounds__(128) void sparse_attn_kernel(
    const float* __restrict__ q, const float* __restrict__ k,
    const float* __restrict__ v, const int* __restrict__ ridx,
    float* __restrict__ out) {
  __shared__ _Float16 smem[6 * TILEH];
  _Float16* Qs = smem;
  _Float16* Ps = smem + 5 * TILEH;

  const int qb   = blockIdx.x;
  const int bh   = blockIdx.y;
  const int tid  = threadIdx.x;
  const int lane = tid & 31;
  const int wave = tid >> 5;
  const int hi   = lane >> 4;
  const int nidx = lane & 15;   // this lane's query column within the wave's 16

  const long base = (long)bh * (4096L * 64L);

  int list0[6] = {0, 0, 0, 0, 0, 0};
  int nkb;
  if (qb == 0) {
    nkb = 64;                            // dense: top queries attend to everything
  } else {
    const int n = qb - 1;
    list0[0] = 0;                        // global block
    list0[1] = 1 + ((n + 62) % 63);      // window (mod-63 wrap)
    list0[2] = 1 + n;
    list0[3] = 1 + ((n + 1) % 63);
    list0[4] = 1 + ridx[2 * n];          // random blocks
    list0[5] = 1 + ridx[2 * n + 1];
    nkb = 6;
  }

  // ---- prologue: Q tile + first K/V tile ----
  {
    const float* qsrc = q + base + (long)qb * 64 * 64;
    float4 rq[8];
    fetch_tileK(rq, qsrc, tid);
    store_tileK(Qs, rq, tid, 0.125f);    // 1/sqrt(64)
  }
  float4 rk[8], rv[8];
  {
    const int kb0 = (qb == 0) ? 0 : list0[0];
    fetch_tileK(rk, k + base + (long)kb0 * 64 * 64, tid);
    fetch_tileV(rv, v + base + (long)kb0 * 64 * 64, tid);
    store_tileK(smem + 1 * TILEH, rk, tid, 1.0f);
    store_tileV(smem + 3 * TILEH, rv, tid);
  }
  __syncthreads();

  // Q as B-matrix fragments (K-dim = d, split 32+32), constant for whole kernel.
  const v16h qb0 = ld_fragB(Qs, wave * 16, 0, lane);
  const v16h qb1 = ld_fragB(Qs, wave * 16, 32, lane);

  float mrow = -__builtin_inff();
  float lrow = 0.f;
  v8f acc[4];                            // O^T tiles: row d = t*16 + r + 8*hi, col = query nidx
  const v8f zeroV = {};
#pragma unroll
  for (int t = 0; t < 4; ++t) acc[t] = zeroV;

  _Float16* Pw = Ps + wave * (16 * LDSTR);

  for (int it = 0; it < nkb; ++it) {
    const int cur = it & 1;
    const _Float16* Ks = smem + (1 + cur) * TILEH;
    const _Float16* Vt = smem + (3 + cur) * TILEH;

    // Issue next tile's global loads now (no wait) so they overlap the compute below.
    const bool havenext = (it + 1 < nkb);
    if (havenext) {
      const int kbn = (qb == 0) ? (it + 1) : list0[it + 1];
      fetch_tileK(rk, k + base + (long)kbn * 64 * 64, tid);
      fetch_tileV(rv, v + base + (long)kbn * 64 * 64, tid);
    }
    // Warm L2 two blocks ahead (global_prefetch_b8, one 128B line per thread).
    if (it + 2 < nkb) {
      const int kb2 = (qb == 0) ? (it + 2) : list0[it + 2];
      __builtin_prefetch(k + base + (long)kb2 * 64 * 64 + tid * 32, 0, 1);
      __builtin_prefetch(v + base + (long)kb2 * 64 * 64 + tid * 32, 0, 1);
    }

    // ---- S^T = K Q^T : 4 key subtiles (M), 1 query tile (N), K-dim = d 32+32 ----
    v8f st[4];
#pragma unroll
    for (int t = 0; t < 4; ++t) {
      v16h a0 = ld_fragA(Ks, t * 16, 0, lane);
      v16h a1 = ld_fragA(Ks, t * 16, 32, lane);
      v8f c = {};
      c = __builtin_amdgcn_wmma_f32_16x16x32_f16(false, a0, false, qb0, (short)0, c, false, false);
      c = __builtin_amdgcn_wmma_f32_16x16x32_f16(false, a1, false, qb1, (short)0, c, false, false);
      st[t] = c;   // st[t][r] = S[query nidx][key t*16 + 8*hi + r]
    }

    // ---- online softmax: per-lane (one query) stats + single xor-16 combine ----
    float tmax = -__builtin_inff();
#pragma unroll
    for (int t = 0; t < 4; ++t)
#pragma unroll
      for (int r = 0; r < 8; ++r) tmax = fmaxf(tmax, st[t][r]);
    tmax = fmaxf(tmax, __shfl_xor(tmax, 16, 32));

    const float mn     = fmaxf(mrow, tmax);
    const float scalef = __expf(mrow - mn);
    mrow = mn;

    float psum = 0.f;
#pragma unroll
    for (int t = 0; t < 4; ++t)
#pragma unroll
      for (int r = 0; r < 8; ++r) {
        float p = __expf(st[t][r] - mn);
        st[t][r] = p;
        psum += p;
      }
    psum += __shfl_xor(psum, 16, 32);
    lrow = lrow * scalef + psum;
#pragma unroll
    for (int t = 0; t < 4; ++t)
#pragma unroll
      for (int r = 0; r < 8; ++r) acc[t][r] *= scalef;

    // ---- P -> LDS row-major [query][key]: 8 consecutive keys per lane per tile = b128 ----
#pragma unroll
    for (int t = 0; t < 4; ++t) {
      Pack8 pk;
#pragma unroll
      for (int r = 0; r < 8; ++r) pk.h[r] = (_Float16)st[t][r];
      *reinterpret_cast<uint4*>(Pw + nidx * LDSTR + t * 16 + 8 * hi) = pk.u;
    }

    // P as B-matrix fragments (K-dim = key, split 32+32); reads are contiguous rows.
    v16h pb0 = ld_fragB(Pw, 0, 0, lane);
    v16h pb1 = ld_fragB(Pw, 0, 32, lane);

    // ---- O^T += V^T P^T : 4 d subtiles (M), K-dim = keys 32+32 ----
#pragma unroll
    for (int t = 0; t < 4; ++t) {
      v16h a0 = ld_fragA(Vt, t * 16, 0, lane);
      v16h a1 = ld_fragA(Vt, t * 16, 32, lane);
      acc[t] = __builtin_amdgcn_wmma_f32_16x16x32_f16(false, a0, false, pb0, (short)0, acc[t], false, false);
      acc[t] = __builtin_amdgcn_wmma_f32_16x16x32_f16(false, a1, false, pb1, (short)0, acc[t], false, false);
    }

    // ---- drain next tile into the other buffer; one barrier covers both hazards ----
    if (havenext) {
      store_tileK(smem + (1 + (1 - cur)) * TILEH, rk, tid, 1.0f);
      store_tileV(smem + (3 + (1 - cur)) * TILEH, rv, tid);
    }
    __syncthreads();
  }

  // ---- epilogue: out[query][d] = acc / l ; 8 consecutive d per tile -> 2 b128 stores ----
  const float rl = 1.0f / lrow;
  float* orow = out + base + (long)(qb * 64 + wave * 16 + nidx) * 64;
#pragma unroll
  for (int t = 0; t < 4; ++t) {
    float4 o0, o1;
    o0.x = acc[t][0] * rl; o0.y = acc[t][1] * rl; o0.z = acc[t][2] * rl; o0.w = acc[t][3] * rl;
    o1.x = acc[t][4] * rl; o1.y = acc[t][5] * rl; o1.z = acc[t][6] * rl; o1.w = acc[t][7] * rl;
    *reinterpret_cast<float4*>(orow + t * 16 + 8 * hi)     = o0;
    *reinterpret_cast<float4*>(orow + t * 16 + 8 * hi + 4) = o1;
  }
}

extern "C" void kernel_launch(void* const* d_in, const int* in_sizes, int n_in,
                              void* d_out, int out_size, void* d_ws, size_t ws_size,
                              hipStream_t stream) {
  const float* q    = (const float*)d_in[0];
  const float* k    = (const float*)d_in[1];
  const float* v    = (const float*)d_in[2];
  const int*   ridx = (const int*)d_in[3];
  float* out = (float*)d_out;

  const int BH = in_sizes[0] / (4096 * 64);  // B*H = 64
  dim3 grid(64, BH);
  dim3 block(128);                           // 4 waves
  sparse_attn_kernel<<<grid, block, 0, stream>>>(q, k, v, ridx, out);
}
